// DeTok_5677946766018
// MI455X (gfx1250) — matrix operand order repaired and checked by reading, not compiled, not run on hardware
//
#include <hip/hip_runtime.h>
#include <hip/hip_bf16.h>

// ---------------------------------------------------------------------------
// Types / helpers
// ---------------------------------------------------------------------------
typedef __attribute__((ext_vector_type(16))) __bf16 v16bf;
typedef __attribute__((ext_vector_type(8)))  float  v8f;
typedef unsigned short ushort_t;

struct U32x8 { uint4 a, b; };

static __device__ inline v16bf mk_v16bf(uint4 lo, uint4 hi) {
    U32x8 t; t.a = lo; t.b = hi;
    return __builtin_bit_cast(v16bf, t);
}

static __device__ inline ushort_t f2bf(float f) {
    unsigned int u = __float_as_uint(f);
    unsigned int r = (u + 0x7FFFu + ((u >> 16) & 1u)) >> 16;
    return (ushort_t)r;
}

#define WMMA_BF16(a, b, c) \
    __builtin_amdgcn_wmma_f32_16x16x32_bf16(false, (a), false, (b), (short)0, (c), false, false)

// ---------------------------------------------------------------------------
// CDNA5 async global->LDS path (ASYNCcnt). Guarded so host pass / older
// toolchains fall back to synchronous VGPR staging.
// Builtin prototype (from hipcc diagnostic): (v4i AS1*, v4i AS3*, Ii, Ii)
// ---------------------------------------------------------------------------
#if defined(__has_builtin)
#if __has_builtin(__builtin_amdgcn_global_load_async_to_lds_b128)
#define ASYNC_LDS 1
#endif
#endif
#ifndef ASYNC_LDS
#define ASYNC_LDS 0
#endif

#if ASYNC_LDS
typedef int v4i_vs __attribute__((vector_size(16)));
typedef __attribute__((address_space(1))) v4i_vs as1_v4i;
typedef __attribute__((address_space(3))) v4i_vs as3_v4i;
static __device__ inline void stage16(const ushort_t* g, ushort_t* l) {
    __builtin_amdgcn_global_load_async_to_lds_b128((as1_v4i*)g, (as3_v4i*)l, 0, 0);
}
static __device__ inline void async_wait4() {
#if __has_builtin(__builtin_amdgcn_s_wait_asynccnt)
    __builtin_amdgcn_s_wait_asynccnt(4);
#else
    asm volatile("s_wait_asynccnt 0x4" ::: "memory");
#endif
}
static __device__ inline void async_wait0() {
#if __has_builtin(__builtin_amdgcn_s_wait_asynccnt)
    __builtin_amdgcn_s_wait_asynccnt(0);
#else
    asm volatile("s_wait_asynccnt 0x0" ::: "memory");
#endif
}
#else
static __device__ inline void stage16(const ushort_t* g, ushort_t* l) {
    *(uint4*)l = *(const uint4*)g;
}
static __device__ inline void async_wait4() {}
static __device__ inline void async_wait0() {}
#endif

// ---------------------------------------------------------------------------
// Model constants
// ---------------------------------------------------------------------------
#define BSZ   64
#define SEQ   256
#define WID   768
#define NLAY  12
#define NHEAD 12
#define HDIM  64
#define HID   2048
#define MROWS (BSZ * SEQ)   // 16384
#define EPSV  1e-6f

// ---------------------------------------------------------------------------
// Weight transpose + fp32 -> bf16 convert:  dst[n*K+k] = bf16(src[k*N+n])
// ---------------------------------------------------------------------------
__global__ void k_transpose_bf16(const float* __restrict__ src,
                                 ushort_t* __restrict__ dst, int K, int N) {
    int gid = blockIdx.x * 256 + threadIdx.x;
    int total = K * N;
    if (gid >= total) return;
    int k = gid / N, n = gid - k * N;
    dst[(size_t)n * K + k] = f2bf(src[(size_t)k * N + n]);
}

// ---------------------------------------------------------------------------
// Embed: z = rms(z_lat @ W_embed + b_embed + pos_emb, ln_pre_w). One block/row.
// ---------------------------------------------------------------------------
__global__ __launch_bounds__(256)
void k_embed(const float* __restrict__ zlat, const float* __restrict__ pos,
             const float* __restrict__ Wemb, const float* __restrict__ bemb,
             const float* __restrict__ lnw, float* __restrict__ z) {
    __shared__ float zl[16];
    __shared__ float red[8];
    int row = blockIdx.x;               // b*256+n
    int n   = row & 255;
    int tid = threadIdx.x, lane = tid & 31, wv = tid >> 5;
    if (tid < 16) zl[tid] = zlat[(size_t)row * 16 + tid];
    __syncthreads();
    float vals[3];
    float ss = 0.f;
#pragma unroll
    for (int i = 0; i < 3; i++) {
        int c = tid + i * 256;
        float acc = bemb[c] + pos[(size_t)n * WID + c];
#pragma unroll
        for (int k = 0; k < 16; k++) acc += zl[k] * Wemb[k * WID + c];
        vals[i] = acc;
        ss += acc * acc;
    }
#pragma unroll
    for (int d = 16; d; d >>= 1) ss += __shfl_xor(ss, d, 32);
    if (lane == 0) red[wv] = ss;
    __syncthreads();
    if (tid == 0) { float t = 0; for (int i = 0; i < 8; i++) t += red[i]; red[0] = t; }
    __syncthreads();
    float r = rsqrtf(red[0] * (1.0f / WID) + EPSV);
#pragma unroll
    for (int i = 0; i < 3; i++) {
        int c = tid + i * 256;
        z[(size_t)row * WID + c] = vals[i] * r * lnw[c];
    }
}

// ---------------------------------------------------------------------------
// RMSNorm fp32 -> bf16. One block per row of 768.
// ---------------------------------------------------------------------------
__global__ __launch_bounds__(256)
void k_rmsnorm(const float* __restrict__ x, const float* __restrict__ w,
               ushort_t* __restrict__ out) {
    __shared__ float red[8];
    int row = blockIdx.x;
    int tid = threadIdx.x, lane = tid & 31, wv = tid >> 5;
    const float* xr = x + (size_t)row * WID;
    float vals[3];
    float ss = 0.f;
#pragma unroll
    for (int i = 0; i < 3; i++) {
        float v = xr[tid + i * 256];
        vals[i] = v; ss += v * v;
    }
#pragma unroll
    for (int d = 16; d; d >>= 1) ss += __shfl_xor(ss, d, 32);
    if (lane == 0) red[wv] = ss;
    __syncthreads();
    if (tid == 0) { float t = 0; for (int i = 0; i < 8; i++) t += red[i]; red[0] = t; }
    __syncthreads();
    float r = rsqrtf(red[0] * (1.0f / WID) + EPSV);
#pragma unroll
    for (int i = 0; i < 3; i++) {
        int c = tid + i * 256;
        out[(size_t)row * WID + c] = f2bf(vals[i] * r * w[c]);
    }
}

// ---------------------------------------------------------------------------
// Tiled bf16 WMMA GEMM.  C(MxN,f32) = A(MxK,bf16) @ BT(NxK,bf16)^T + bias[n]
// EPI==1: also add resid (fp32 MxN; may alias C).
// Block: 256 thr = 8 waves as 4(M)x2(N); wave tile 32x64; block tile 128x128.
// Double-buffered staging through async global->LDS (ASYNCcnt pipeline).
// ---------------------------------------------------------------------------
#define TK 32
#define LSTR 48   // padded ushorts per LDS row (96B, 16B-aligned sublanes)

template <int EPI>
__global__ __launch_bounds__(256)
void k_gemm_bf16(const ushort_t* __restrict__ A, const ushort_t* __restrict__ BT,
                 const float* __restrict__ bias, const float* __restrict__ resid,
                 float* __restrict__ C, int M, int N, int K) {
    __shared__ __align__(16) ushort_t sA[2][128 * LSTR];
    __shared__ __align__(16) ushort_t sB[2][128 * LSTR];

    const int tid  = threadIdx.x;
    const int wv   = tid >> 5;
    const int lane = tid & 31;
    const int l15  = lane & 15;
    const bool hiL = lane >= 16;
    const int bm = blockIdx.y * 128;
    const int bn = blockIdx.x * 128;
    const int wm = (wv >> 1) * 32;   // wave row base inside block tile
    const int wn = (wv & 1) * 64;    // wave col base inside block tile

    v8f acc[2][4];
#pragma unroll
    for (int mt = 0; mt < 2; mt++)
#pragma unroll
        for (int nt = 0; nt < 4; nt++)
#pragma unroll
            for (int g = 0; g < 8; g++) acc[mt][nt][g] = 0.f;

    const int sr = tid >> 1;          // staging row 0..127
    const int sc = (tid & 1) * 16;    // staging col 0 / 16
    const ushort_t* gaBase = A  + (size_t)(bm + sr) * K + sc;
    const ushort_t* gbBase = BT + (size_t)(bn + sr) * K + sc;
    const int soff = sr * LSTR + sc;

    // 4 stage ops per thread per tile (each 16B)
    // prologue: tile 0 into buffer 0
    stage16(gaBase,     &sA[0][soff]);
    stage16(gaBase + 8, &sA[0][soff + 8]);
    stage16(gbBase,     &sB[0][soff]);
    stage16(gbBase + 8, &sB[0][soff + 8]);

    int buf = 0;
    for (int k0 = 0; k0 < K; k0 += TK) {
        const bool hasNext = (k0 + TK) < K;
        if (hasNext) {
            stage16(gaBase + k0 + TK,     &sA[buf ^ 1][soff]);
            stage16(gaBase + k0 + TK + 8, &sA[buf ^ 1][soff + 8]);
            stage16(gbBase + k0 + TK,     &sB[buf ^ 1][soff]);
            stage16(gbBase + k0 + TK + 8, &sB[buf ^ 1][soff + 8]);
            if (k0 + 2 * TK < K) {  // warm L2 two tiles ahead
                __builtin_prefetch(gaBase + k0 + 2 * TK, 0, 0);
                __builtin_prefetch(gbBase + k0 + 2 * TK, 0, 0);
            }
            async_wait4();   // current tile landed; next still in flight
        } else {
            async_wait0();
        }
        __syncthreads();

        v16bf av[2], bv[4];
#pragma unroll
        for (int mt = 0; mt < 2; mt++) {
            int r = wm + mt * 16 + l15;
            int ko = hiL ? 8 : 0;  // A frag: lanes>=16 hold K 8..15 / 24..31
            uint4 lo = *(const uint4*)&sA[buf][r * LSTR + ko];
            uint4 hi = *(const uint4*)&sA[buf][r * LSTR + ko + 16];
            av[mt] = mk_v16bf(lo, hi);
        }
#pragma unroll
        for (int nt = 0; nt < 4; nt++) {
            int c = wn + nt * 16 + l15;
            int ko = hiL ? 8 : 0;  // B frag: lanes>=16 hold K 8..15 / 24..31
            uint4 lo = *(const uint4*)&sB[buf][c * LSTR + ko];
            uint4 hi = *(const uint4*)&sB[buf][c * LSTR + ko + 16];
            bv[nt] = mk_v16bf(lo, hi);
        }
#pragma unroll
        for (int mt = 0; mt < 2; mt++)
#pragma unroll
            for (int nt = 0; nt < 4; nt++)
                acc[mt][nt] = WMMA_BF16(av[mt], bv[nt], acc[mt][nt]);
        __syncthreads();
        buf ^= 1;
    }

#pragma unroll
    for (int mt = 0; mt < 2; mt++)
#pragma unroll
        for (int nt = 0; nt < 4; nt++) {
            int col = bn + wn + nt * 16 + l15;
            float bc = bias[col];
#pragma unroll
            for (int g = 0; g < 8; g++) {
                int row = bm + wm + mt * 16 + g + (hiL ? 8 : 0);
                size_t o = (size_t)row * N + col;
                float v = acc[mt][nt][g] + bc;
                if (EPI == 1) v += resid[o];
                C[o] = v;
            }
        }
}

// ---------------------------------------------------------------------------
// RoPE + repack: qkv f32 [B*N][3*768] -> q,k bf16 [BH][N][D], vT bf16 [BH][D][N]
// ---------------------------------------------------------------------------
__global__ void k_rope_pack(const float* __restrict__ qkv, const float* __restrict__ rope,
                            ushort_t* __restrict__ qb, ushort_t* __restrict__ kb,
                            ushort_t* __restrict__ vT) {
    int t = blockIdx.x * 256 + threadIdx.x;
    if (t >= BSZ * NHEAD * SEQ * (HDIM / 2)) return;
    int dp = t & 31;
    int n  = (t >> 5) & 255;
    int h  = (t >> 13) % NHEAD;
    int b  = t / (32 * 256 * NHEAD);
    int d0 = dp * 2, d1 = d0 + 1;

    size_t row = (size_t)(b * SEQ + n) * (3 * WID);
    int cb = h * HDIM;
    float c0 = rope[n * 128 + d0],       c1 = rope[n * 128 + d1];
    float s0 = rope[n * 128 + 64 + d0],  s1 = rope[n * 128 + 64 + d1];
    float q0 = qkv[row + cb + d0],           q1 = qkv[row + cb + d1];
    float k0 = qkv[row + WID + cb + d0],     k1 = qkv[row + WID + cb + d1];
    float v0 = qkv[row + 2 * WID + cb + d0], v1 = qkv[row + 2 * WID + cb + d1];

    float qa = q0 * c0 - q1 * s0, qc = q1 * c1 + q0 * s1;
    float ka = k0 * c0 - k1 * s0, kc = k1 * c1 + k0 * s1;

    size_t ob = ((size_t)(b * NHEAD + h) * SEQ + n) * HDIM;
    qb[ob + d0] = f2bf(qa); qb[ob + d1] = f2bf(qc);
    kb[ob + d0] = f2bf(ka); kb[ob + d1] = f2bf(kc);
    size_t vb = ((size_t)(b * NHEAD + h) * HDIM + d0) * SEQ + n;
    vT[vb]       = f2bf(v0);
    vT[vb + SEQ] = f2bf(v1);
}

// ---------------------------------------------------------------------------
// Fused flash attention. One block per (b,h); 8 waves; wave = 32 q rows.
// S = q@k^T via WMMA; online softmax; P repacked via wave-private LDS; o += P@v.
// Writes o (transposed to token-major) as bf16 into obuf [B*N][768].
// ---------------------------------------------------------------------------
__global__ __launch_bounds__(256)
void k_attn(const ushort_t* __restrict__ qb, const ushort_t* __restrict__ kb,
            const ushort_t* __restrict__ vT, ushort_t* __restrict__ obuf) {
    __shared__ __align__(16) ushort_t sP[8 * 32 * LSTR];  // per-wave P tiles

    const int bh = blockIdx.x;
    const int b = bh / NHEAD, h = bh % NHEAD;
    const int tid = threadIdx.x, wv = tid >> 5, lane = tid & 31;
    const int l15 = lane & 15;
    const bool hiL = lane >= 16;
    const float scale = 0.125f;  // 1/sqrt(64)

    const ushort_t* qbase = qb + (size_t)bh * SEQ * HDIM;
    const ushort_t* kbase = kb + (size_t)bh * SEQ * HDIM;
    const ushort_t* vbase = vT + (size_t)bh * HDIM * SEQ;

    // q fragments: [msub][ksub] ; A-frag addressing
    v16bf aq[2][2];
#pragma unroll
    for (int mt = 0; mt < 2; mt++)
#pragma unroll
        for (int kt = 0; kt < 2; kt++) {
            int n  = wv * 32 + mt * 16 + l15;
            int ko = kt * 32 + (hiL ? 8 : 0);
            const ushort_t* p = qbase + (size_t)n * HDIM + ko;
            aq[mt][kt] = mk_v16bf(*(const uint4*)p, *(const uint4*)(p + 16));
        }

    v8f o[2][4];
    float rmax[2][8], rsum[2][8];
#pragma unroll
    for (int mt = 0; mt < 2; mt++) {
#pragma unroll
        for (int dt = 0; dt < 4; dt++)
#pragma unroll
            for (int g = 0; g < 8; g++) o[mt][dt][g] = 0.f;
#pragma unroll
        for (int g = 0; g < 8; g++) { rmax[mt][g] = -1e30f; rsum[mt][g] = 0.f; }
    }

    for (int j = 0; j < 8; j++) {
        const int m0 = j * 32;
        // --- S = q @ k^T (k rows are the B^T layout: contiguous per column) ---
        v16bf bk[2][2];
#pragma unroll
        for (int nt = 0; nt < 2; nt++)
#pragma unroll
            for (int kt = 0; kt < 2; kt++) {
                int m = m0 + nt * 16 + l15;
                const ushort_t* p = kbase + (size_t)m * HDIM + kt * 32 + (hiL ? 8 : 0);
                bk[nt][kt] = mk_v16bf(*(const uint4*)p, *(const uint4*)(p + 16));
            }
        v8f cs[2][2];
#pragma unroll
        for (int mt = 0; mt < 2; mt++)
#pragma unroll
            for (int nt = 0; nt < 2; nt++) {
#pragma unroll
                for (int g = 0; g < 8; g++) cs[mt][nt][g] = 0.f;
#pragma unroll
                for (int kt = 0; kt < 2; kt++)
                    cs[mt][nt] = WMMA_BF16(aq[mt][kt], bk[nt][kt], cs[mt][nt]);
            }
        // --- online softmax; write P (bf16) to wave-private LDS ---
#pragma unroll
        for (int mt = 0; mt < 2; mt++) {
#pragma unroll
            for (int g = 0; g < 8; g++) {
                float s0 = cs[mt][0][g] * scale;
                float s1 = cs[mt][1][g] * scale;
                float mx = fmaxf(s0, s1);
#pragma unroll
                for (int d = 8; d; d >>= 1) mx = fmaxf(mx, __shfl_xor(mx, d, 32));
                float nm = fmaxf(rmax[mt][g], mx);
                float al = __expf(rmax[mt][g] - nm);
                rmax[mt][g] = nm;
                float p0 = __expf(s0 - nm);
                float p1 = __expf(s1 - nm);
                float ps = p0 + p1;
#pragma unroll
                for (int d = 8; d; d >>= 1) ps += __shfl_xor(ps, d, 32);
                rsum[mt][g] = rsum[mt][g] * al + ps;
#pragma unroll
                for (int dt = 0; dt < 4; dt++) o[mt][dt][g] *= al;
                int pr = mt * 16 + g + (hiL ? 8 : 0);
                sP[wv * 32 * LSTR + pr * LSTR + l15]      = f2bf(p0);
                sP[wv * 32 * LSTR + pr * LSTR + 16 + l15] = f2bf(p1);
            }
        }
        // --- P as A fragments (LDS same-wave ops are in-order) ---
        v16bf ap[2];
#pragma unroll
        for (int mt = 0; mt < 2; mt++) {
            int base = wv * 32 * LSTR + (mt * 16 + l15) * LSTR + (hiL ? 8 : 0);
            ap[mt] = mk_v16bf(*(const uint4*)&sP[base], *(const uint4*)&sP[base + 16]);
        }
        // --- vT B fragments: column d contiguous over m ---
        v16bf bv[4];
#pragma unroll
        for (int dt = 0; dt < 4; dt++) {
            int d = dt * 16 + l15;
            const ushort_t* p = vbase + (size_t)d * SEQ + m0 + (hiL ? 8 : 0);
            bv[dt] = mk_v16bf(*(const uint4*)p, *(const uint4*)(p + 16));
        }
#pragma unroll
        for (int mt = 0; mt < 2; mt++)
#pragma unroll
            for (int dt = 0; dt < 4; dt++)
                o[mt][dt] = WMMA_BF16(ap[mt], bv[dt], o[mt][dt]);
    }

    // --- normalize & write o back token-major as bf16 ---
#pragma unroll
    for (int mt = 0; mt < 2; mt++)
#pragma unroll
        for (int dt = 0; dt < 4; dt++) {
            int col = h * HDIM + dt * 16 + l15;
#pragma unroll
            for (int g = 0; g < 8; g++) {
                int n = wv * 32 + mt * 16 + g + (hiL ? 8 : 0);
                float v = o[mt][dt][g] / rsum[mt][g];
                obuf[(size_t)(b * SEQ + n) * WID + col] = f2bf(v);
            }
        }
}

// ---------------------------------------------------------------------------
// SwiGLU: hmid = silu(x1) * x2  (x12 f32 Mx4096 -> bf16 Mx2048)
// ---------------------------------------------------------------------------
__global__ void k_swiglu(const float* __restrict__ x12, ushort_t* __restrict__ hmid) {
    size_t gid = (size_t)blockIdx.x * 256 + threadIdx.x;
    if (gid >= (size_t)MROWS * HID) return;
    size_t row = gid / HID;
    int c = (int)(gid - row * HID);
    float x1 = x12[row * (2 * HID) + c];
    float x2 = x12[row * (2 * HID) + HID + c];
    float sig = 1.0f / (1.0f + __expf(-x1));
    hmid[gid] = f2bf(x1 * sig * x2);
}

// ---------------------------------------------------------------------------
// Unpatchify + 3x3 SAME conv.  y f32 [B*N][768] -> out f32 [B][3][256][256]
// ---------------------------------------------------------------------------
__global__ void k_unpatch_conv(const float* __restrict__ y, const float* __restrict__ cw,
                               const float* __restrict__ cb, float* __restrict__ out) {
    size_t gid = (size_t)blockIdx.x * 256 + threadIdx.x;
    const size_t total = (size_t)BSZ * 3 * 256 * 256;
    if (gid >= total) return;
    int x  = (int)(gid & 255);
    int yy = (int)((gid >> 8) & 255);
    int co = (int)((gid >> 16) % 3);
    int b  = (int)(gid / ((size_t)3 * 65536));
    float acc = cb[co];
#pragma unroll
    for (int ci = 0; ci < 3; ci++)
#pragma unroll
        for (int ky = 0; ky < 3; ky++)
#pragma unroll
            for (int kx = 0; kx < 3; kx++) {
                int iy = yy + ky - 1, ix = x + kx - 1;
                if (iy < 0 || iy > 255 || ix < 0 || ix > 255) continue;
                int g1 = iy >> 4, p1 = iy & 15, g2 = ix >> 4, p2 = ix & 15;
                size_t row = (size_t)b * SEQ + g1 * 16 + g2;
                int col = (p1 * 16 + p2) * 3 + ci;
                acc += cw[((co * 3 + ci) * 3 + ky) * 3 + kx] * y[row * WID + col];
            }
    out[gid] = acc;
}

// ---------------------------------------------------------------------------
// Host orchestration
// ---------------------------------------------------------------------------
extern "C" void kernel_launch(void* const* d_in, const int* in_sizes, int n_in,
                              void* d_out, int out_size, void* d_ws, size_t ws_size,
                              hipStream_t stream) {
    const float* z_lat   = (const float*)d_in[0];
    const float* pos     = (const float*)d_in[1];
    const float* Wemb    = (const float*)d_in[2];
    const float* bemb    = (const float*)d_in[3];
    const float* lnpre   = (const float*)d_in[4];
    const float* n1_w    = (const float*)d_in[5];
    const float* qkv_w   = (const float*)d_in[6];
    const float* qkv_b   = (const float*)d_in[7];
    const float* proj_w  = (const float*)d_in[8];
    const float* proj_b  = (const float*)d_in[9];
    const float* n2_w    = (const float*)d_in[10];
    const float* w12_w   = (const float*)d_in[11];
    const float* w12_b   = (const float*)d_in[12];
    const float* w3_w    = (const float*)d_in[13];
    const float* w3_b    = (const float*)d_in[14];
    const float* lnpost  = (const float*)d_in[15];
    const float* conv1_w = (const float*)d_in[16];
    const float* conv1_b = (const float*)d_in[17];
    const float* convo_w = (const float*)d_in[18];
    const float* convo_b = (const float*)d_in[19];
    const float* rope    = (const float*)d_in[20];

    // ---- workspace bump allocator (256B aligned) ----
    char* wsp = (char*)d_ws;
    auto alloc = [&](size_t bytes) -> void* {
        void* p = (void*)wsp;
        wsp += (bytes + 255) & ~(size_t)255;
        return p;
    };
    ushort_t* qkvT  = (ushort_t*)alloc((size_t)NLAY * 2304 * 768 * 2);
    ushort_t* projT = (ushort_t*)alloc((size_t)NLAY * 768 * 768 * 2);
    ushort_t* w12T  = (ushort_t*)alloc((size_t)NLAY * 4096 * 768 * 2);
    ushort_t* w3T   = (ushort_t*)alloc((size_t)NLAY * 768 * 2048 * 2);
    ushort_t* c1T   = (ushort_t*)alloc((size_t)768 * 768 * 2);
    float*    z     = (float*)alloc((size_t)MROWS * WID * 4);
    ushort_t* hbuf  = (ushort_t*)alloc((size_t)MROWS * WID * 2);
    float*    qkvb  = (float*)alloc((size_t)MROWS * 3 * WID * 4);
    ushort_t* qr    = (ushort_t*)alloc((size_t)BSZ * NHEAD * SEQ * HDIM * 2);
    ushort_t* kr    = (ushort_t*)alloc((size_t)BSZ * NHEAD * SEQ * HDIM * 2);
    ushort_t* vTb   = (ushort_t*)alloc((size_t)BSZ * NHEAD * SEQ * HDIM * 2);
    ushort_t* obuf  = (ushort_t*)alloc((size_t)MROWS * WID * 2);
    float*    x12   = (float*)alloc((size_t)MROWS * 2 * HID * 4);
    ushort_t* hmid  = (ushort_t*)alloc((size_t)MROWS * HID * 2);
    float*    ybuf  = (float*)alloc((size_t)MROWS * WID * 4);

    // ---- 1) weight convert + transpose to bf16 (done every call; deterministic) ----
    for (int l = 0; l < NLAY; l++) {
        int e;
        e = 768 * 2304;
        k_transpose_bf16<<<(e + 255) / 256, 256, 0, stream>>>(
            qkv_w + (size_t)l * e, qkvT + (size_t)l * e, 768, 2304);
        e = 768 * 768;
        k_transpose_bf16<<<(e + 255) / 256, 256, 0, stream>>>(
            proj_w + (size_t)l * e, projT + (size_t)l * e, 768, 768);
        e = 768 * 4096;
        k_transpose_bf16<<<(e + 255) / 256, 256, 0, stream>>>(
            w12_w + (size_t)l * e, w12T + (size_t)l * e, 768, 4096);
        e = 2048 * 768;
        k_transpose_bf16<<<(e + 255) / 256, 256, 0, stream>>>(
            w3_w + (size_t)l * e, w3T + (size_t)l * e, 2048, 768);
    }
    k_transpose_bf16<<<(768 * 768 + 255) / 256, 256, 0, stream>>>(conv1_w, c1T, 768, 768);

    // ---- 2) embed + pre-LN ----
    k_embed<<<MROWS, 256, 0, stream>>>(z_lat, pos, Wemb, bemb, lnpre, z);

    // ---- 3) transformer layers ----
    const int ropeN = BSZ * NHEAD * SEQ * (HDIM / 2);
    for (int l = 0; l < NLAY; l++) {
        k_rmsnorm<<<MROWS, 256, 0, stream>>>(z, n1_w + (size_t)l * WID, hbuf);
        k_gemm_bf16<0><<<dim3(2304 / 128, MROWS / 128), 256, 0, stream>>>(
            hbuf, qkvT + (size_t)l * 2304 * 768, qkv_b + (size_t)l * 2304, nullptr,
            qkvb, MROWS, 2304, 768);
        k_rope_pack<<<(ropeN + 255) / 256, 256, 0, stream>>>(qkvb, rope, qr, kr, vTb);
        k_attn<<<BSZ * NHEAD, 256, 0, stream>>>(qr, kr, vTb, obuf);
        k_gemm_bf16<1><<<dim3(768 / 128, MROWS / 128), 256, 0, stream>>>(
            obuf, projT + (size_t)l * 768 * 768, proj_b + (size_t)l * WID, z,
            z, MROWS, 768, 768);
        k_rmsnorm<<<MROWS, 256, 0, stream>>>(z, n2_w + (size_t)l * WID, hbuf);
        k_gemm_bf16<0><<<dim3(4096 / 128, MROWS / 128), 256, 0, stream>>>(
            hbuf, w12T + (size_t)l * 4096 * 768, w12_b + (size_t)l * 4096, nullptr,
            x12, MROWS, 4096, 768);
        size_t sw = (size_t)MROWS * HID;
        k_swiglu<<<(unsigned)((sw + 255) / 256), 256, 0, stream>>>(x12, hmid);
        k_gemm_bf16<1><<<dim3(768 / 128, MROWS / 128), 256, 0, stream>>>(
            hmid, w3T + (size_t)l * 768 * 2048, w3_b + (size_t)l * WID, z,
            z, MROWS, 768, 2048);
    }

    // ---- 4) post-LN, patch projection, unpatchify + 3x3 conv ----
    k_rmsnorm<<<MROWS, 256, 0, stream>>>(z, lnpost, hbuf);
    k_gemm_bf16<0><<<dim3(768 / 128, MROWS / 128), 256, 0, stream>>>(
        hbuf, c1T, conv1_b, nullptr, ybuf, MROWS, 768, 768);
    size_t tot = (size_t)BSZ * 3 * 256 * 256;
    k_unpatch_conv<<<(unsigned)((tot + 255) / 256), 256, 0, stream>>>(
        ybuf, convo_w, convo_b, (float*)d_out);
}
